// RSNN_with_delay_synapse_88845693485075
// MI455X (gfx1250) — matrix operand order repaired and checked by reading, not compiled, not run on hardware
//
#include <hip/hip_runtime.h>

// RSNN with delay synapses — persistent WMMA kernel for gfx1250 (wave32).
// B=128, T=250, NIN=700, H=1024, NOUT=20, D=5.
#define Bb   128
#define Tn   250
#define NINP 700
#define KX   704      // NIN padded to multiple of 32
#define Hh   1024
#define NO   20
#define DD   5
#define TAUc 0.96f
#define VTHc 1.0f

typedef _Float16 v16h __attribute__((ext_vector_type(16)));
typedef _Float16 v8h  __attribute__((ext_vector_type(8)));
typedef float    v8f  __attribute__((ext_vector_type(8)));

static __device__ inline v8f zero8() {
  v8f z = {0.f,0.f,0.f,0.f,0.f,0.f,0.f,0.f};
  return z;
}

// A-matrix fragment (16x32 f16) from row-major [16][S] halfword tile in LDS.
// Layout: lanes 0-15 -> M=lane, K = k0+0..15; lanes 16-31 -> M=lane-16, K = k0+16..31 (matches B).
static __device__ inline v16h load_a_lds(const _Float16* base, int S, int k0, int lane) {
  int m    = lane & 15;
  int koff = (lane < 16) ? 0 : 16;
  const _Float16* p = base + m * S + k0 + koff;
  v8h lo = *(const v8h*)(p);        // K + 0..7   (16B aligned)
  v8h hi = *(const v8h*)(p + 8);    // K + 8..15  (16B aligned)
  v16h r;
  #pragma unroll
  for (int i = 0; i < 8; ++i) { r[i] = lo[i]; r[8 + i] = hi[i]; }
  return r;
}

// B-matrix fragment (32x16 f16) from N-major / K-contiguous global weights [N][Ks].
// Layout (contiguous K, matches A): lanes 0-15 -> col n0+lane, K=k0..k0+15; lanes 16-31 -> col n0+lane-16, K=k0+16..k0+31.
static __device__ inline v16h load_b_glb(const _Float16* __restrict__ base, int Ks, int n0, int k0, int lane) {
  int n = n0 + (lane & 15);
  int k = k0 + ((lane >> 4) << 4);
  return *(const v16h*)(base + (size_t)n * Ks + k);   // 32 contiguous bytes
}

// ---------------- prep kernels: fp16 weight repacking ----------------

__global__ void prep_wdel(const float* __restrict__ rec, const float* __restrict__ mask,
                          _Float16* __restrict__ wdelh) {
  size_t i = (size_t)blockIdx.x * 256 + threadIdx.x;   // over D*H*H
  if (i >= (size_t)DD * Hh * Hh) return;
  int d   = (int)(i / ((size_t)Hh * Hh));
  int rem = (int)(i % ((size_t)Hh * Hh));
  int j   = rem / Hh;    // N index (output col)
  int ii  = rem % Hh;    // K index (input row)
  float w = rec[(size_t)ii * Hh + j] * mask[((size_t)ii * Hh + j) * DD + d];
  wdelh[i] = (_Float16)w;  // wdelh[d][n=j][k=i]
}

__global__ void prep_fc1(const float* __restrict__ w, _Float16* __restrict__ o) {
  int i = blockIdx.x * 256 + threadIdx.x;   // over H*KX
  if (i >= Hh * KX) return;
  int j = i / KX, k = i % KX;
  o[i] = (_Float16)((k < NINP) ? w[(size_t)j * NINP + k] : 0.f);  // [n=j][k], zero-padded K
}

__global__ void prep_outw(const float* __restrict__ w, _Float16* __restrict__ o) {
  int i = blockIdx.x * 256 + threadIdx.x;   // over 32*H
  if (i >= 32 * Hh) return;
  int oo = i / Hh, j = i % Hh;
  o[i] = (_Float16)((oo < NO) ? w[(size_t)oo * Hh + j] : 0.f);    // [n=o][k=j], rows 20..31 zero
}

// ---------------- persistent RSNN kernel ----------------
// Grid: 8 workgroups (one per 16-row batch tile), 256 threads = 8 waves.
// Wave w owns output columns [w*128, w*128+128) = 8 WMMA N-tiles.
__global__ __launch_bounds__(256, 1) void rsnn_persistent(
    const float*    __restrict__ x,      // [B][T][700]
    const _Float16* __restrict__ fc1h,   // [1024][704]
    const _Float16* __restrict__ wdelh,  // [5][1024][1024]
    const _Float16* __restrict__ outwh,  // [32][1024]
    float*          __restrict__ out) {  // [B][T][20]
  __shared__ _Float16 ring[DD][16][Hh];  // spike history ring, slot (t%5) holds sp(t)   (160 KB)
  __shared__ _Float16 xh[16][KX];        // current-step input tile, fp16, zero-padded   (22 KB)

  const int tid  = threadIdx.x;
  const int lane = tid & 31;
  const int wave = tid >> 5;
  const int b0   = blockIdx.x * 16;

  // zero spike history
  for (int i = tid; i < DD * 16 * Hh; i += 256) ((_Float16*)ring)[i] = (_Float16)0.f;

  // hidden membrane state in registers, mapped 1:1 onto WMMA C/D fragments
  v8f memh[8];
  #pragma unroll
  for (int nt = 0; nt < 8; ++nt) memh[nt] = zero8();
  // output membrane state (used by wave 0 only)
  v8f memo0 = zero8(), memo1 = zero8();

  __syncthreads();

  for (int t = 0; t < Tn; ++t) {
    // ---- stage x_t into LDS as fp16 (cooperative) ----
    for (int i = tid; i < 16 * KX; i += 256) {
      int m = i / KX, k = i % KX;
      float v = (k < NINP) ? x[((size_t)(b0 + m) * Tn + t) * NINP + k] : 0.f;
      xh[m][k] = (_Float16)v;
    }
    __syncthreads();   // xh ready; prev-step spike writes visible

    const int slot_w = t % DD;   // slot to be overwritten with sp(t)

    v8f acc[8];
    #pragma unroll
    for (int nt = 0; nt < 8; ++nt) acc[nt] = zero8();

    // ---- input GEMM: x_t[16,704] @ fc1^T -> acc ----
    for (int k0 = 0; k0 < KX; k0 += 32) {
      v16h a = load_a_lds(&xh[0][0], KX, k0, lane);
      #pragma unroll
      for (int nt = 0; nt < 8; ++nt) {
        int n0 = wave * 128 + nt * 16;
        v16h b = load_b_glb(fc1h, KX, n0, k0, lane);
        acc[nt] = __builtin_amdgcn_wmma_f32_16x16x32_f16(false, a, false, b,
                                                         (short)0, acc[nt], false, false);
      }
    }

    // ---- recurrent delay taps: tap d uses sp(t-(D-d)) = ring slot (t+d)%D ----
    for (int d = 0; d < DD; ++d) {
      int slot = (t + d) % DD;
      const _Float16* wd = wdelh + (size_t)d * Hh * Hh;
      for (int k0 = 0; k0 < Hh; k0 += 32) {
        v16h a = load_a_lds(&ring[slot][0][0], Hh, k0, lane);
        #pragma unroll
        for (int nt = 0; nt < 8; ++nt) {
          int n0 = wave * 128 + nt * 16;
          v16h b = load_b_glb(wd, Hh, n0, k0, lane);
          acc[nt] = __builtin_amdgcn_wmma_f32_16x16x32_f16(false, a, false, b,
                                                           (short)0, acc[nt], false, false);
        }
      }
    }
    __syncthreads();   // all ring reads (incl. slot t%5 = sp(t-5)) complete before overwrite

    // ---- LIF hidden update + write sp(t) into ring slot ----
    {
      const int mbase = (lane < 16) ? 0 : 8;
      const int ncol  = lane & 15;
      #pragma unroll
      for (int nt = 0; nt < 8; ++nt) {
        int n = wave * 128 + nt * 16 + ncol;
        #pragma unroll
        for (int r = 0; r < 8; ++r) {
          float prev  = memh[nt][r];
          float decay = (prev >= VTHc) ? 0.f : (TAUc * prev);   // soft reset: TAU*mem*(1-sp_prev)
          float m     = decay + acc[nt][r];
          memh[nt][r] = m;
          ring[slot_w][mbase + r][n] = (_Float16)((m >= VTHc) ? 1.f : 0.f);
        }
      }
    }
    __syncthreads();   // sp(t) fully written

    // ---- output layer (wave 0): sp(t)[16,1024] @ out_w^T[1024,32] ----
    if (wave == 0) {
      v8f oa0 = zero8(), oa1 = zero8();
      for (int k0 = 0; k0 < Hh; k0 += 32) {
        v16h a  = load_a_lds(&ring[slot_w][0][0], Hh, k0, lane);
        v16h bA = load_b_glb(outwh, Hh, 0,  k0, lane);
        v16h bB = load_b_glb(outwh, Hh, 16, k0, lane);
        oa0 = __builtin_amdgcn_wmma_f32_16x16x32_f16(false, a, false, bA, (short)0, oa0, false, false);
        oa1 = __builtin_amdgcn_wmma_f32_16x16x32_f16(false, a, false, bB, (short)0, oa1, false, false);
      }
      const int mbase = (lane < 16) ? 0 : 8;
      const int ncol  = lane & 15;
      #pragma unroll
      for (int r = 0; r < 8; ++r) {
        int b = b0 + mbase + r;
        { // cols 0..15 (all < NO? no: NO=20, cols 0..15 valid)
          float prev = memo0[r];
          float dec  = (prev >= VTHc) ? 0.f : (TAUc * prev);
          float mm   = dec + oa0[r];
          memo0[r]   = mm;
          out[((size_t)b * Tn + t) * NO + ncol] = (mm >= VTHc) ? 1.f : 0.f;
        }
        { // cols 16..31, only 16..19 stored
          float prev = memo1[r];
          float dec  = (prev >= VTHc) ? 0.f : (TAUc * prev);
          float mm   = dec + oa1[r];
          memo1[r]   = mm;
          int o = 16 + ncol;
          if (o < NO) out[((size_t)b * Tn + t) * NO + o] = (mm >= VTHc) ? 1.f : 0.f;
        }
      }
    }
  }
}

extern "C" void kernel_launch(void* const* d_in, const int* in_sizes, int n_in,
                              void* d_out, int out_size, void* d_ws, size_t ws_size,
                              hipStream_t stream) {
  const float* x      = (const float*)d_in[0];  // [B][T][700]
  const float* fc1_w  = (const float*)d_in[1];  // [H][700]
  const float* recur  = (const float*)d_in[2];  // [H][H]
  const float* out_w  = (const float*)d_in[3];  // [20][H]
  const float* dmask  = (const float*)d_in[4];  // [H][H][D]
  float* out = (float*)d_out;                   // [B][T][20]

  // workspace layout (fp16 repacked weights), 256B-aligned
  char* ws = (char*)d_ws;
  size_t off = 0;
  _Float16* wdelh = (_Float16*)(ws + off); off += (size_t)DD * Hh * Hh * 2;  // 10.0 MB
  off = (off + 255) & ~(size_t)255;
  _Float16* fc1h  = (_Float16*)(ws + off); off += (size_t)Hh * KX * 2;       // 1.4 MB
  off = (off + 255) & ~(size_t)255;
  _Float16* outwh = (_Float16*)(ws + off); off += (size_t)32 * Hh * 2;       // 64 KB
  (void)ws_size; (void)in_sizes; (void)n_in; (void)out_size;

  {
    size_t n = (size_t)DD * Hh * Hh;
    prep_wdel<<<dim3((unsigned)((n + 255) / 256)), dim3(256), 0, stream>>>(recur, dmask, wdelh);
  }
  prep_fc1 <<<dim3((Hh * KX + 255) / 256), dim3(256), 0, stream>>>(fc1_w, fc1h);
  prep_outw<<<dim3((32 * Hh + 255) / 256), dim3(256), 0, stream>>>(out_w, outwh);

  rsnn_persistent<<<dim3(Bb / 16), dim3(256), 0, stream>>>(x, fc1h, wdelh, outwh, out);
}
